// RnnInferenceNetwork_58179626992322
// MI455X (gfx1250) — compile-verified, matrix operand order checked by reading, not run on hardware
//
#include <hip/hip_runtime.h>
#include <cstddef>

// ---------------- types ----------------
typedef __attribute__((ext_vector_type(16))) __bf16   v16bf;
typedef __attribute__((ext_vector_type(8)))  float    v8f;
typedef __attribute__((ext_vector_type(4)))  unsigned int u32x4;

// ---------------- problem constants ----------------
#define BATCH   256
#define TSTEPS  200
#define OBS     128
#define HID     128
#define LAT     32
#define NSAMP   50
#define BSEQ    (BATCH * NSAMP)      // 12800 sequences
#define KTOT    (OBS + LAT + HID)    // 288 = concat(x, z, h)
#define MTILE   32                   // sequences per workgroup
#define NTHREADS 256                 // 8 waves (wave32)

// fragment-ordered weight storage: 1 frag = 16x16x32-bf16 B operand = 32 lanes * 32B
#define FRAG_ELEMS 512               // bf16 elems per fragment (1 KB)
#define NGFRAG  (32 * 9)             // 512 gate cols / 16  x  288 K / 32
#define NMUFRAG (2 * 4)              // 32 lat cols / 16    x  128 K / 32
#define WFRAGS_PER_WAVE 36           // 4 gates * 9 kt, contiguous per wave

#define ASTR    296                  // padded A-buffer row stride in bf16 (288 + 8)

// LDS layout (units: bf16 elements)
#define WG_OFF   0
#define WM_OFF   (NGFRAG * FRAG_ELEMS)                 // 147456
#define AB_OFF   (WM_OFF + NMUFRAG * FRAG_ELEMS)       // 151552
#define MISC_OFF (AB_OFF + MTILE * ASTR)               // 161024  (byte 322048)
#define SMEM_BYTES (MISC_OFF * 2 + 256)                // 322304 B  (< 320KB WGP LDS)

#define SBTL ((size_t)NSAMP * BATCH * TSTEPS * LAT)    // 81,920,000
#define SBT  ((size_t)NSAMP * BATCH * TSTEPS)          //  2,560,000
#define OFF_ENT (SBTL)
#define OFF_LP  (SBTL + SBT)

__constant__ float kLOG2PI = 1.8378770664093453f;

// ---------------- helpers ----------------
__device__ __forceinline__ v8f wmma_bf16(v16bf a, v16bf b, v8f c) {
  // D = A(16x32 bf16) x B(32x16 bf16) + C(f32)
  return __builtin_amdgcn_wmma_f32_16x16x32_bf16(false, a, false, b,
                                                 (short)0, c, false, false);
}

// B fragment: chunk0 at p, chunk1 at p+256 elems (both 16B, immediate-offsettable)
__device__ __forceinline__ v16bf ld_bfrag(const __bf16* p) {
  union { u32x4 u[2]; v16bf v; } r;
  r.u[0] = *(const u32x4*)p;          // ds_load_b128  offset:(frag*1024)
  r.u[1] = *(const u32x4*)(p + 256);  // ds_load_b128  offset:(frag*1024+512)
  return r.v;
}

// A fragment: chunk0 at p, chunk1 at p+16 elems.
// lane<16: K = kbase+{0..7,16..23}; lane>=16: K = kbase+{8..15,24..31}; M = lane%16.
__device__ __forceinline__ v16bf ld_afrag(const __bf16* p) {
  union { u32x4 u[2]; v16bf v; } r;
  r.u[0] = *(const u32x4*)p;
  r.u[1] = *(const u32x4*)(p + 16);
  return r.v;
}

__device__ __forceinline__ float sigm(float x) { return 1.0f / (1.0f + __expf(-x)); }
__device__ __forceinline__ float tanh_e(float x) {
  float e = __expf(2.0f * x);
  return (e - 1.0f) / (e + 1.0f);
}

// ---------------- kernel ----------------
__global__ __launch_bounds__(NTHREADS)
void lstm_vi_persistent(const float* __restrict__ x,     // [B,T,OBS]
                        const float* __restrict__ eps,   // [T,BS,L]
                        const float* __restrict__ Wih,   // [4H, OBS+L]
                        const float* __restrict__ Whh,   // [4H, H]
                        const float* __restrict__ bih,   // [4H]
                        const float* __restrict__ bhh,   // [4H]
                        const float* __restrict__ Wmu,   // [L, H]
                        const float* __restrict__ bmu,   // [L]
                        const float* __restrict__ cov,   // [L]
                        float* __restrict__ out)
{
  extern __shared__ __bf16 smem[];
  __bf16* Wg   = smem + WG_OFF;
  __bf16* Wm   = smem + WM_OFF;
  __bf16* Ab   = smem + AB_OFF;              // [MTILE][ASTR] = [x | z | h] per row
  float*  misc = (float*)(smem + MISC_OFF);  // [0..31]=std, [32]=logdet

  const int tid  = threadIdx.x;
  const int wave = tid >> 5;
  const int lane = tid & 31;
  const int l15  = lane & 15;
  const int hi16 = (lane & 16) ? 8 : 0;
  const int bs0  = blockIdx.x * MTILE;

  // ---- init: convert weights fp32 -> bf16 fragments in LDS ----
  // Gate frag id = (wave*4 + g)*9 + kt  -> each wave's 36 frags are contiguous (36KB).
  for (int j = tid; j < (NGFRAG + NMUFRAG) * 64; j += NTHREADS) {
    const int frag  = j >> 6;
    const int chunk = (j >> 5) & 1;
    const int ln    = j & 31;
    const int nsub  = ln & 15;
    const int koff  = chunk * 16 + ((ln & 16) ? 8 : 0);
    alignas(16) __bf16 tmp[8];
    if (frag < NGFRAG) {
      const int wv  = frag / WFRAGS_PER_WAVE;        // owning wave
      const int rem = frag % WFRAGS_PER_WAVE;
      const int g   = rem / 9;
      const int kt  = rem % 9;
      const int n   = (g * 8 + wv) * 16 + nsub;      // gate column
      const int k0  = kt * 32 + koff;
#pragma unroll
      for (int e = 0; e < 8; ++e) {
        const int k = k0 + e;   // 8-runs never straddle 160 (160 % 8 == 0)
        const float w = (k < OBS + LAT) ? Wih[n * (OBS + LAT) + k]
                                        : Whh[n * HID + (k - (OBS + LAT))];
        tmp[e] = (__bf16)w;
      }
      *(u32x4*)(Wg + frag * FRAG_ELEMS + chunk * 256 + ln * 8) = *(const u32x4*)tmp;
    } else {
      const int f2 = frag - NGFRAG;                  // [nt][kt], nt-major
      const int nt = f2 >> 2, kt = f2 & 3;
      const int n  = nt * 16 + nsub;
      const int k0 = kt * 32 + koff;
#pragma unroll
      for (int e = 0; e < 8; ++e) tmp[e] = (__bf16)Wmu[n * HID + k0 + e];
      *(u32x4*)(Wm + f2 * FRAG_ELEMS + chunk * 256 + ln * 8) = *(const u32x4*)tmp;
    }
  }
  // zero A buffer (h0 = c0 = z0 = 0)
  for (int j = tid; j < MTILE * ASTR; j += NTHREADS) Ab[j] = (__bf16)0.0f;
  // std + logdet
  if (tid < LAT) misc[tid] = sqrtf(cov[tid]);
  if (tid == NTHREADS - 1) {
    float s = 0.0f;
    for (int l = 0; l < LAT; ++l) s += __logf(cov[l]);
    misc[LAT] = s;
  }
  __syncthreads();

  // ---- per-role constants ----
  const int jcol = wave * 16 + l15;       // this lane's hidden-unit column
  float biasg[4];
#pragma unroll
  for (int g = 0; g < 4; ++g) biasg[g] = bih[g * HID + jcol] + bhh[g * HID + jcol];

  const float logdet = misc[LAT];
  const float ent_c  = 0.5f * (float)LAT * (1.0f + kLOG2PI) + 0.5f * logdet;
  int mt_mu = 0, nt_mu = 0, lcol = 0;
  float stdl = 0.0f, bmul = 0.0f;
  if (wave < 4) {
    mt_mu = wave >> 1;  nt_mu = wave & 1;
    lcol  = nt_mu * 16 + l15;
    stdl  = misc[lcol];
    bmul  = bmu[lcol];
  }

  // ---- loop-invariant LDS base pointers (all per-step offsets are immediates) ----
  const __bf16* wgw = Wg + wave * (WFRAGS_PER_WAVE * FRAG_ELEMS) + lane * 8;
  const __bf16* wmw = Wm + (nt_mu * 4) * FRAG_ELEMS + lane * 8;
  const __bf16* a0b = Ab + l15 * ASTR + hi16;          // mt=0 A-frag base
  const __bf16* a1b = a0b + 16 * ASTR;                 // mt=1 A-frag base
  const __bf16* amb = Ab + (mt_mu * 16 + l15) * ASTR + hi16 + (OBS + LAT); // mean A base
  __bf16* hwb = Ab + hi16 * ASTR + (OBS + LAT) + jcol; // h scatter base
  __bf16* zwb = Ab + (mt_mu * 16 + hi16) * ASTR + OBS + lcol; // z scatter base

  // stage x_t into A-buffer K[0..128) as bf16 (waves 6,7)
  auto load_x = [&](int t) {
    const int idx  = tid - 192;            // 0..63
    const int row  = idx >> 1;
    const int half = idx & 1;
    const int b    = (bs0 + row) & (BATCH - 1);   // sample-major tiling: b = bs % B
    const float* srcf = x + ((size_t)b * TSTEPS + t) * OBS + half * 64;
    if (t + 1 < TSTEPS) __builtin_prefetch(srcf + OBS, 0, 0);  // global_prefetch_b8
    const float4* src = (const float4*)srcf;
    __bf16* dst = Ab + row * ASTR + half * 64;
#pragma unroll
    for (int q = 0; q < 16; ++q) {
      float4 v = src[q];
      union { __bf16 h[4]; unsigned long long u; } pk;
      pk.h[0] = (__bf16)v.x; pk.h[1] = (__bf16)v.y;
      pk.h[2] = (__bf16)v.z; pk.h[3] = (__bf16)v.w;
      *(unsigned long long*)(dst + q * 4) = pk.u;
    }
  };
  if (wave >= 6) load_x(0);
  __syncthreads();

  // cell state (f32, register-resident, C-matrix layout)
  v8f creg[2];
#pragma unroll
  for (int m = 0; m < 2; ++m)
#pragma unroll
    for (int v = 0; v < 8; ++v) creg[m][v] = 0.0f;

  // ================= time loop =================
  for (int t = 0; t < TSTEPS; ++t) {
    // --- phase A: gates = [x|z|h] @ Wcat^T + bias  (72 WMMA / wave) ---
    v8f acc[4][2];
#pragma unroll
    for (int g = 0; g < 4; ++g)
#pragma unroll
      for (int m = 0; m < 2; ++m)
#pragma unroll
        for (int v = 0; v < 8; ++v) acc[g][m][v] = biasg[g];

#pragma unroll
    for (int kt = 0; kt < 9; ++kt) {
      v16bf a0 = ld_afrag(a0b + kt * 32);
      v16bf a1 = ld_afrag(a1b + kt * 32);
#pragma unroll
      for (int g = 0; g < 4; ++g) {
        v16bf bf = ld_bfrag(wgw + (g * 9 + kt) * FRAG_ELEMS); // imm offsets < 36KB
        acc[g][0] = wmma_bf16(a0, bf, acc[g][0]);
        acc[g][1] = wmma_bf16(a1, bf, acc[g][1]);
      }
    }

    // --- phase B: LSTM cell (register-local, i/f/g/o co-resident per lane) ---
    float hreg[2][8];
#pragma unroll
    for (int m = 0; m < 2; ++m)
#pragma unroll
      for (int v = 0; v < 8; ++v) {
        const float iv = sigm(acc[0][m][v]);
        const float fv = sigm(acc[1][m][v]);
        const float gv = tanh_e(acc[2][m][v]);
        const float ov = sigm(acc[3][m][v]);
        const float cn = fv * creg[m][v] + iv * gv;
        creg[m][v] = cn;
        hreg[m][v] = ov * tanh_e(cn);
      }

    __syncthreads();   // all reads of A-buffer (x,z,h of step t) complete

    // --- phase C: scatter h_new (bf16) into A-buffer K[160..288) ---
#pragma unroll
    for (int m = 0; m < 2; ++m)
#pragma unroll
      for (int v = 0; v < 8; ++v)
        hwb[(m * 16 + v) * ASTR] = (__bf16)hreg[m][v];   // ds_store_b16 offset:imm

    __syncthreads();   // h_new visible

    // --- phase D: mean+sample / logprob / entropy / x staging (wave-split) ---
    if (wave < 4) {
      // mean tile [16x16]: 4 WMMA over K=128 of h
      v8f mv;
#pragma unroll
      for (int v = 0; v < 8; ++v) mv[v] = bmul;
#pragma unroll
      for (int kt = 0; kt < 4; ++kt) {
        v16bf a  = ld_afrag(amb + kt * 32);
        v16bf bf = ld_bfrag(wmw + kt * FRAG_ELEMS);
        mv = wmma_bf16(a, bf, mv);
      }
      const int rbase = mt_mu * 16 + hi16;
      const float* ep = eps + ((size_t)t * BSEQ + bs0 + rbase) * LAT + lcol;
      float* zo = out + (size_t)(bs0 + rbase) * TSTEPS * LAT + (size_t)t * LAT + lcol;
#pragma unroll
      for (int v = 0; v < 8; ++v) {
        const float e = ep[(size_t)v * LAT];
        const float z = mv[v] + e * stdl;
        zo[(size_t)v * TSTEPS * LAT] = z;                 // sampled_s
        zwb[v * ASTR] = (__bf16)z;                        // z_prev carry (imm offset)
      }
    } else if (wave == 4) {
      // log N(z; mean, diag) = -0.5*(||eps||^2 + L*log2pi + logdet)
      const int r = lane;                                 // rows 0..31
      const float4* ep = (const float4*)(eps + ((size_t)t * BSEQ + bs0 + r) * LAT);
      float s = 0.0f;
#pragma unroll
      for (int q = 0; q < 8; ++q) {
        float4 e4 = ep[q];
        s += e4.x * e4.x + e4.y * e4.y + e4.z * e4.z + e4.w * e4.w;
      }
      out[OFF_LP + (size_t)(bs0 + r) * TSTEPS + t] =
          -0.5f * (s + (float)LAT * kLOG2PI + logdet);
    } else if (wave == 5) {
      out[OFF_ENT + (size_t)(bs0 + lane) * TSTEPS + t] = ent_c;
    } else {
      if (t + 1 < TSTEPS) load_x(t + 1);
    }

    __syncthreads();   // z_t and x_{t+1} staged for next iteration
  }
}

// ---------------- host launch ----------------
extern "C" void kernel_launch(void* const* d_in, const int* in_sizes, int n_in,
                              void* d_out, int out_size, void* d_ws, size_t ws_size,
                              hipStream_t stream) {
  (void)in_sizes; (void)n_in; (void)out_size; (void)d_ws; (void)ws_size;
  const float* x    = (const float*)d_in[0];
  const float* eps  = (const float*)d_in[1];
  const float* Wih  = (const float*)d_in[2];
  const float* Whh  = (const float*)d_in[3];
  const float* bih  = (const float*)d_in[4];
  const float* bhh  = (const float*)d_in[5];
  const float* Wmu  = (const float*)d_in[6];
  const float* bmu  = (const float*)d_in[7];
  const float* cov  = (const float*)d_in[8];
  float* out = (float*)d_out;

  (void)hipFuncSetAttribute((const void*)lstm_vi_persistent,
                            hipFuncAttributeMaxDynamicSharedMemorySize, SMEM_BYTES);

  lstm_vi_persistent<<<BSEQ / MTILE, NTHREADS, SMEM_BYTES, stream>>>(
      x, eps, Wih, Whh, bih, bhh, Wmu, bmu, cov, out);
}
